// NN_with_leaf_emb_80513456931490
// MI455X (gfx1250) — matrix operand order repaired and checked by reading, not compiled, not run on hardware
//
#include <hip/hip_runtime.h>
#include <hip/hip_bf16.h>
#include <stdint.h>

// ---------------- problem constants (from reference) ----------------
#define N_NODES  1000000
#define FEAT_DIM 22
#define N_EST    16
#define N_LEAVES 31
#define LEAF_DIM 64
#define NEDGE    500000

#define WAVES  4                 // waves per block (128 threads)
#define NTILES (NEDGE / 16)      // 31250 tiles of 16 edges

// WMMA fragment staging geometry: each (ntile,kblock) = 32 lanes * 8 dwords
#define L1_NT 3
#define L1_KB 4
#define L2_NT 2
#define L2_KB 2
#define L1_OFF 0
#define L2_OFF (L1_OFF + L1_NT * L1_KB * 256)   // 3072
#define L3_OFF (L2_OFF + L2_NT * L2_KB * 256)   // 4096
#define L4_OFF (L3_OFF + 256)                   // 4352
#define WFRAG_TOTAL (L4_OFF + 256)              // 4608 dwords = 18 KB

typedef __attribute__((ext_vector_type(16))) __bf16 v16bf;
typedef __attribute__((ext_vector_type(8)))  float  v8f;
static_assert(sizeof(v16bf) == 32, "v16bf must be 8 VGPRs");

// ---------------- helpers ----------------
__device__ __forceinline__ unsigned int bf16rne(float f) {
  unsigned int u = __float_as_uint(f);
  unsigned int r = ((u >> 16) & 1u) + 0x7FFFu;   // round-to-nearest-even
  return (u + r) >> 16;
}
__device__ __forceinline__ unsigned int packbf(float a, float b) {
  return bf16rne(a) | (bf16rne(b) << 16);
}

// A-fragment (16-bit 16x32): lane<16 -> K = {base..base+7, base+16..base+23},
// lane>=16 -> base += 8.  `row` points at the lane's M row in LDS (bf16 units).
__device__ __forceinline__ v16bf ldfragA(const unsigned short* row, int base) {
  union { uint4 q[2]; v16bf v; } f;
  f.q[0] = *(const uint4*)(row + base);
  f.q[1] = *(const uint4*)(row + base + 16);
  return f.v;
}
// B-fragment: pre-swizzled contiguous 8 dwords per lane.
__device__ __forceinline__ v16bf ldfragB(const unsigned int* p) {
  union { uint4 q[2]; v16bf v; } f;
  f.q[0] = *(const uint4*)(p);
  f.q[1] = *(const uint4*)(p + 4);
  return f.v;
}

// Stage W (fi x fo, row-major) into B-fragment layout, bf16, zero-padded.
// dst[i]: i = ((nt*kblocks + kb)*32 + lane)*8 + j ; packed K = kb*32 + half*8 + (j>>2)*16 + 2(j&3) (+1)
__device__ __forceinline__ void stage_frags(const float* __restrict__ W,
                                            int fi, int fo, int kblocks,
                                            unsigned int* dst, int total,
                                            int tid, int nthr) {
  for (int i = tid; i < total; i += nthr) {
    int j    = i & 7;
    int lane = (i >> 3) & 31;
    int blk  = i >> 8;
    int kb   = blk % kblocks;
    int nt   = blk / kblocks;
    int col  = nt * 16 + (lane & 15);
    int k    = kb * 32 + (lane >> 4) * 8 + (j >> 2) * 16 + 2 * (j & 3);
    float f0 = (k     < fi && col < fo) ? W[(size_t)k       * fo + col] : 0.f;
    float f1 = (k + 1 < fi && col < fo) ? W[(size_t)(k + 1) * fo + col] : 0.f;
    dst[i] = packbf(f0, f1);
  }
}

#define WMMA_BF16(A, B, C) \
  __builtin_amdgcn_wmma_f32_16x16x32_bf16(false, (A), false, (B), (short)0, (C), false, false)

// ---------------- fused kernel ----------------
__global__ __launch_bounds__(128) void nn_leaf_emb_fused(
    const float* __restrict__ X,  const float* __restrict__ LT,
    const float* __restrict__ W1, const float* __restrict__ b1,
    const float* __restrict__ W2, const float* __restrict__ b2,
    const float* __restrict__ W3, const float* __restrict__ b3,
    const float* __restrict__ W4, const float* __restrict__ b4,
    const int* __restrict__ ED,   const int* __restrict__ PL,
    float* __restrict__ out)
{
  __shared__ __align__(16) unsigned int   s_wfrag[WFRAG_TOTAL];     // 18 KB
  __shared__ float                        s_bias[72];
  __shared__ __align__(16) unsigned short s_bufA[WAVES][16 * 136];  // feats K=128 (+8 pad)
  __shared__ __align__(16) unsigned short s_bufB[WAVES][16 * 72];   // h K<=64 (+8 pad)

  const int tid = threadIdx.x;

  // ---- per-block: stage all weights as bf16 B-fragments + biases ----
  stage_frags(W1, 106, 36, L1_KB, s_wfrag + L1_OFF, L1_NT * L1_KB * 256, tid, 128);
  stage_frags(W2,  36, 20, L2_KB, s_wfrag + L2_OFF, L2_NT * L2_KB * 256, tid, 128);
  stage_frags(W3,  20, 12, 1,     s_wfrag + L3_OFF, 256, tid, 128);
  stage_frags(W4,  12,  1, 1,     s_wfrag + L4_OFF, 256, tid, 128);
  for (int i = tid; i < 69; i += 128) {
    float v;
    if      (i < 36) v = b1[i];
    else if (i < 56) v = b2[i - 36];
    else if (i < 68) v = b3[i - 56];
    else             v = b4[0];
    s_bias[i] = v;
  }
  __syncthreads();

  const int  wid  = tid >> 5;
  const int  lane = tid & 31;
  const int  half = lane >> 4;      // 0: lanes 0-15, 1: lanes 16-31
  const int  e    = lane & 15;      // edge-in-tile == WMMA row M (lane-pair shared)
  const long tile = (long)blockIdx.x * WAVES + wid;
  if (tile >= NTILES) return;

  unsigned short* bufA = s_bufA[wid];
  unsigned short* bufB = s_bufB[wid];

  // Pre-zero bufB so uninitialized LDS (possible NaN bit patterns) never feeds a WMMA.
  {
    unsigned int* p = (unsigned int*)bufB;
    #pragma unroll
    for (int i = lane; i < 16 * 72 / 2; i += 32) p[i] = 0u;
  }

  // ---- feature build: 16 edges x 128 bf16 (cols 106..127 zero) ----
  const long gedge = tile * 16 + e;
  const int  chan = ED[gedge * 3 + 0];
  const int  dev  = ED[gedge * 3 + 1];
  const int  lab  = ED[gedge * 3 + 2];
  unsigned int* rowu = (unsigned int*)(bufA + e * 136);

  if (half == 0) {   // chan_data -> cols 0..19 ; zero cols 106..115
    const float2* xc = (const float2*)(X + (size_t)chan * FEAT_DIM);
    #pragma unroll
    for (int q = 0; q < 10; ++q) { float2 v = xc[q]; rowu[q] = packbf(v.x, v.y); }
    #pragma unroll
    for (int q = 53; q < 58; ++q) rowu[q] = 0u;
  } else {           // dev_data -> cols 20..41 ; zero cols 116..127
    const float2* xd = (const float2*)(X + (size_t)dev * FEAT_DIM);
    #pragma unroll
    for (int q = 0; q < 11; ++q) { float2 v = xd[q]; rowu[10 + q] = packbf(v.x, v.y); }
    #pragma unroll
    for (int q = 58; q < 64; ++q) rowu[q] = 0u;
  }

  // leaf mean: each half accumulates 32 of 64 dims over 16 estimators
  float lacc[32];
  #pragma unroll
  for (int q = 0; q < 32; ++q) lacc[q] = 0.f;
  const int* pl = PL + gedge * N_EST;
  for (int t = 0; t < N_EST; ++t) {
    int li = pl[t];
    const float4* lp = (const float4*)(LT + (((size_t)(t * N_LEAVES + li)) << 6) + (half << 5));
    #pragma unroll
    for (int q = 0; q < 8; ++q) {
      float4 v = lp[q];
      lacc[4 * q + 0] += v.x; lacc[4 * q + 1] += v.y;
      lacc[4 * q + 2] += v.z; lacc[4 * q + 3] += v.w;
    }
  }
  {
    const float s = 1.0f / 16.0f;
    int c0 = (42 + 32 * half) >> 1;   // dword index: cols 42..73 / 74..105
    #pragma unroll
    for (int q = 0; q < 16; ++q) rowu[c0 + q] = packbf(lacc[2 * q] * s, lacc[2 * q + 1] * s);
  }
  __builtin_amdgcn_wave_barrier();    // LDS is per-wave in-order; fence the scheduler

  const v8f z8 = {0.f, 0.f, 0.f, 0.f, 0.f, 0.f, 0.f, 0.f};
  const int n = lane & 15;            // WMMA output column within N-tile

  // ---- layer 1: feats(16x128) @ W1(128x48) -> h1(16x36) : 12 WMMAs ----
  v8f c1[3] = {z8, z8, z8};
  const unsigned short* arow = bufA + e * 136;
  #pragma unroll
  for (int kb = 0; kb < L1_KB; ++kb) {
    v16bf a = ldfragA(arow, kb * 32 + half * 8);
    #pragma unroll
    for (int nt = 0; nt < L1_NT; ++nt) {
      v16bf b = ldfragB(s_wfrag + L1_OFF + ((nt * L1_KB + kb) * 32 + lane) * 8);
      c1[nt] = WMMA_BF16(a, b, c1[nt]);
    }
  }
  #pragma unroll
  for (int nt = 0; nt < L1_NT; ++nt) {
    int col = nt * 16 + n;
    if (col < 36) {
      float bi = s_bias[col];
      #pragma unroll
      for (int r = 0; r < 8; ++r) {
        float v = c1[nt][r] + bi;           // D: M = r + 8*half, N = col
        v = v > 0.f ? v : 0.f;
        bufB[(r + half * 8) * 72 + col] = (unsigned short)bf16rne(v);
      }
    }
  }
  __builtin_amdgcn_wave_barrier();

  // ---- layer 2: h1(16x64) @ W2(64x32) -> h2(16x20) : 4 WMMAs ----
  v8f c2[2] = {z8, z8};
  const unsigned short* brow = bufB + e * 72;
  #pragma unroll
  for (int kb = 0; kb < L2_KB; ++kb) {
    v16bf a = ldfragA(brow, kb * 32 + half * 8);
    #pragma unroll
    for (int nt = 0; nt < L2_NT; ++nt) {
      v16bf b = ldfragB(s_wfrag + L2_OFF + ((nt * L2_KB + kb) * 32 + lane) * 8);
      c2[nt] = WMMA_BF16(a, b, c2[nt]);
    }
  }
  #pragma unroll
  for (int nt = 0; nt < L2_NT; ++nt) {
    int col = nt * 16 + n;
    if (col < 20) {
      float bi = s_bias[36 + col];
      #pragma unroll
      for (int r = 0; r < 8; ++r) {
        float v = c2[nt][r] + bi;
        v = v > 0.f ? v : 0.f;
        bufA[(r + half * 8) * 136 + col] = (unsigned short)bf16rne(v);
      }
    }
  }
  __builtin_amdgcn_wave_barrier();

  // ---- layer 3: h2(16x32) @ W3(32x16) -> h3(16x12) : 1 WMMA ----
  // (bufA cols 20..31 hold stale finite bf16; staged W3 rows >=20 are zero)
  v8f c3 = z8;
  {
    v16bf a = ldfragA(arow, half * 8);
    v16bf b = ldfragB(s_wfrag + L3_OFF + lane * 8);
    c3 = WMMA_BF16(a, b, c3);
  }
  if (n < 12) {
    float bi = s_bias[56 + n];
    #pragma unroll
    for (int r = 0; r < 8; ++r) {
      float v = c3[r] + bi;
      v = v > 0.f ? v : 0.f;
      bufB[(r + half * 8) * 72 + n] = (unsigned short)bf16rne(v);
    }
  }
  __builtin_amdgcn_wave_barrier();

  // ---- layer 4: h3(16x32) @ W4(32x16, col0 valid) -> sigmoid : 1 WMMA ----
  // (bufB cols 12..31 hold stale finite h1 bf16; staged W4 rows >=12 are zero)
  v8f c4 = z8;
  {
    v16bf a = ldfragA(brow, half * 8);
    v16bf b = ldfragB(s_wfrag + L4_OFF + lane * 8);
    c4 = WMMA_BF16(a, b, c4);
  }
  if (n == 0) {
    float bi = s_bias[68];
    #pragma unroll
    for (int r = 0; r < 8; ++r) {
      float v = c4[r] + bi;
      v = 1.f / (1.f + __expf(-v));
      out[tile * 16 + half * 8 + r] = v;    // h output, (E,1) flattened
    }
  }
  if (half == 0) out[(size_t)NEDGE + gedge] = (float)lab;  // labels output
}

// ---------------- host launch ----------------
extern "C" void kernel_launch(void* const* d_in, const int* in_sizes, int n_in,
                              void* d_out, int out_size, void* d_ws, size_t ws_size,
                              hipStream_t stream) {
  const float* X  = (const float*)d_in[0];
  const float* LT = (const float*)d_in[1];
  const float* W1 = (const float*)d_in[2];
  const float* b1 = (const float*)d_in[3];
  const float* W2 = (const float*)d_in[4];
  const float* b2 = (const float*)d_in[5];
  const float* W3 = (const float*)d_in[6];
  const float* b3 = (const float*)d_in[7];
  const float* W4 = (const float*)d_in[8];
  const float* b4 = (const float*)d_in[9];
  const int*   ED = (const int*)d_in[10];
  const int*   PL = (const int*)d_in[11];
  float* out = (float*)d_out;

  int blocks = (NTILES + WAVES - 1) / WAVES;   // 7813 blocks x 128 threads
  nn_leaf_emb_fused<<<dim3(blocks), dim3(WAVES * 32), 0, stream>>>(
      X, LT, W1, b1, W2, b2, W3, b3, W4, b4, ED, PL, out);
}
